// MentionExtractor_90331752170180
// MI455X (gfx1250) — compile-verified, matrix operand order checked by reading, not run on hardware
//
#include <hip/hip_runtime.h>
#include <hip/hip_bf16.h>

#define B_  8
#define L_  2048
#define D_  768
#define S_  512
#define K_  1536   // 2*D
#define N_  768    // D

typedef __attribute__((ext_vector_type(16))) __bf16 v16bf;
typedef __attribute__((ext_vector_type(8)))  float  v8f;
typedef __attribute__((ext_vector_type(8)))  unsigned int v8u;

// round-to-nearest-even f32 -> bf16 (as raw u16)
static __device__ __forceinline__ unsigned short f2bf(float x) {
    unsigned int u = __builtin_bit_cast(unsigned int, x);
    u += 0x7FFFu + ((u >> 16) & 1u);
    return (unsigned short)(u >> 16);
}

// ---------------------------------------------------------------------------
// Kernel 1: W_down [K_ x N_] f32 row-major  ->  Wt [N_ x K_] bf16 row-major
// ---------------------------------------------------------------------------
__global__ __launch_bounds__(256) void wt_kernel(const float* __restrict__ W,
                                                 unsigned short* __restrict__ Wt) {
    int idx = blockIdx.x * 256 + threadIdx.x;     // 0 .. N_*K_-1
    int n = idx / K_;
    int k = idx - n * K_;
    Wt[idx] = f2bf(W[k * N_ + n]);
}

// ---------------------------------------------------------------------------
// Kernel 2: span max/mean pooling -> cat [B_*S_ x K_] bf16
//   one block per span, 256 threads, 3 channels per thread
// ---------------------------------------------------------------------------
__global__ __launch_bounds__(256) void pool_kernel(const float* __restrict__ X,
                                                   const long long* __restrict__ spans,
                                                   unsigned short* __restrict__ cat) {
    int sp = blockIdx.x;                  // 0 .. B_*S_-1
    int b  = sp >> 9;                     // / S_
    long long st = spans[2 * sp + 0];
    long long en = spans[2 * sp + 1];
    int width = (int)(en - st);           // 0..31
    float inv = 1.0f / (float)(width + 1);

    const float* base = X + (long long)b * L_ * D_;
    unsigned short* crow = cat + (long long)sp * K_;

    for (int c = threadIdx.x; c < D_; c += 256) {
        float mx = -1.0e32f;
        float sm = 0.0f;
        for (int r = 0; r <= width; ++r) {
            long long t = en - r;
            if (t < 0) continue;          // matches reference mask semantics
            float v = base[t * D_ + c];
            mx = fmaxf(mx, v);
            sm += v;
        }
        crow[c]       = f2bf(mx);
        crow[D_ + c]  = f2bf(sm * inv);
    }
}

// ---------------------------------------------------------------------------
// Kernel 3: out[M x N_] = cat[M x K_] (bf16) @ Wt^T + bias, M = B_*S_ = 4096
//   wave-per-16x64 strip; v_wmma_f32_16x16x32_bf16, K-loop of 48
// ---------------------------------------------------------------------------
__global__ __launch_bounds__(256) void gemm_kernel(const unsigned short* __restrict__ A,
                                                   const unsigned short* __restrict__ Wt,
                                                   const float* __restrict__ bias,
                                                   float* __restrict__ out) {
    int tid  = threadIdx.x;
    int lane = tid & 31;
    int wid  = blockIdx.x * 8 + (tid >> 5);   // 0 .. 3071

    int mtile  = wid / 12;                    // 0 .. 255
    int ngroup = wid - mtile * 12;            // 0 .. 11 (4 ntiles each)
    int m0 = mtile * 16;
    int n0 = ngroup * 64;

    int row = lane & 15;                      // M (A) / N (B) index within tile
    int kb  = (lane >> 4) << 3;               // 0 or 8 (ISA 16-bit A/B layout)

    // per-lane row pointers, in 32-bit (bf16-pair) units
    const unsigned int* Arow = (const unsigned int*)A + (((size_t)(m0 + row) * K_) >> 1);
    const unsigned int* Brow0 = (const unsigned int*)Wt + (((size_t)(n0 +  0 + row) * K_) >> 1);
    const unsigned int* Brow1 = (const unsigned int*)Wt + (((size_t)(n0 + 16 + row) * K_) >> 1);
    const unsigned int* Brow2 = (const unsigned int*)Wt + (((size_t)(n0 + 32 + row) * K_) >> 1);
    const unsigned int* Brow3 = (const unsigned int*)Wt + (((size_t)(n0 + 48 + row) * K_) >> 1);

    v8f acc0 = {}, acc1 = {}, acc2 = {}, acc3 = {};

    for (int ks = 0; ks < K_; ks += 32) {
        int kbase = (ks + kb) >> 1;           // uint index of first pair

        v8u au;
#pragma unroll
        for (int j = 0; j < 4; ++j) au[j]     = Arow[kbase + j];       // K = kb+2j
#pragma unroll
        for (int j = 0; j < 4; ++j) au[4 + j] = Arow[kbase + 8 + j];   // K = 16+kb+2j
        v16bf a = __builtin_bit_cast(v16bf, au);

        v8u bu;
#pragma unroll
        for (int j = 0; j < 4; ++j) bu[j]     = Brow0[kbase + j];
#pragma unroll
        for (int j = 0; j < 4; ++j) bu[4 + j] = Brow0[kbase + 8 + j];
        acc0 = __builtin_amdgcn_wmma_f32_16x16x32_bf16(false, a, false,
                 __builtin_bit_cast(v16bf, bu), (short)0, acc0, false, false);

#pragma unroll
        for (int j = 0; j < 4; ++j) bu[j]     = Brow1[kbase + j];
#pragma unroll
        for (int j = 0; j < 4; ++j) bu[4 + j] = Brow1[kbase + 8 + j];
        acc1 = __builtin_amdgcn_wmma_f32_16x16x32_bf16(false, a, false,
                 __builtin_bit_cast(v16bf, bu), (short)0, acc1, false, false);

#pragma unroll
        for (int j = 0; j < 4; ++j) bu[j]     = Brow2[kbase + j];
#pragma unroll
        for (int j = 0; j < 4; ++j) bu[4 + j] = Brow2[kbase + 8 + j];
        acc2 = __builtin_amdgcn_wmma_f32_16x16x32_bf16(false, a, false,
                 __builtin_bit_cast(v16bf, bu), (short)0, acc2, false, false);

#pragma unroll
        for (int j = 0; j < 4; ++j) bu[j]     = Brow3[kbase + j];
#pragma unroll
        for (int j = 0; j < 4; ++j) bu[4 + j] = Brow3[kbase + 8 + j];
        acc3 = __builtin_amdgcn_wmma_f32_16x16x32_bf16(false, a, false,
                 __builtin_bit_cast(v16bf, bu), (short)0, acc3, false, false);
    }

    // C/D layout: lane holds column n = row; VGPR v -> M = v + 8*(lane>>4)
    int mrow = m0 + ((lane >> 4) << 3);
#pragma unroll
    for (int g = 0; g < 4; ++g) {
        int n = n0 + g * 16 + row;
        float bv = bias[n];
        v8f acc = (g == 0) ? acc0 : (g == 1) ? acc1 : (g == 2) ? acc2 : acc3;
        float* op = out + (size_t)mrow * N_ + n;
#pragma unroll
        for (int v = 0; v < 8; ++v)
            op[(size_t)v * N_] = acc[v] + bv;
    }
}

// ---------------------------------------------------------------------------
extern "C" void kernel_launch(void* const* d_in, const int* in_sizes, int n_in,
                              void* d_out, int out_size, void* d_ws, size_t ws_size,
                              hipStream_t stream) {
    const float*     X     = (const float*)d_in[0];      // (B,L,D) f32
    const long long* spans = (const long long*)d_in[1];  // (B,S,2) i64
    const float*     W     = (const float*)d_in[2];      // (2D,D)  f32
    const float*     bias  = (const float*)d_in[3];      // (D,)    f32
    float*           out   = (float*)d_out;              // (B,S,D) f32

    unsigned short* cat = (unsigned short*)d_ws;                 // 4096 x 1536 bf16
    unsigned short* Wt  = cat + (size_t)(B_ * S_) * K_;          // 768 x 1536 bf16

    wt_kernel  <<<(N_ * K_) / 256, 256, 0, stream>>>(W, Wt);
    pool_kernel<<<B_ * S_,         256, 0, stream>>>(X, spans, cat);
    gemm_kernel<<<(256 * 12) / 8,  256, 0, stream>>>(cat, Wt, bias, out);
}
